// GraphConv_75685913690234
// MI455X (gfx1250) — matrix operand order repaired and checked by reading, not compiled, not run on hardware
//
#include <hip/hip_runtime.h>
#include <hip/hip_bf16.h>

typedef __attribute__((ext_vector_type(16))) _Float16 v16h;
typedef __attribute__((ext_vector_type(8)))  _Float16 h8;
typedef __attribute__((ext_vector_type(8)))  float    v8f;

#define N_USERS 100000
#define N_ITEMS 150000
#define DIM     64
#define N_EDGES 2000000
#define NNZ     2000000

// ---------------- utility kernels ----------------

__global__ void zero_f32(float* __restrict__ p, int n) {
    int i = blockIdx.x * blockDim.x + threadIdx.x;
    if (i < n) p[i] = 0.0f;
}

__global__ void copy_f32(const float* __restrict__ src, float* __restrict__ dst, int n) {
    int i = blockIdx.x * blockDim.x + threadIdx.x;
    if (i < n) dst[i] = src[i];
}

// degree: cnt[head[e]] += 1
__global__ void degree_kernel(const int* __restrict__ head, float* __restrict__ cnt, int n_edges) {
    int e = blockIdx.x * blockDim.x + threadIdx.x;
    if (e < n_edges) atomicAdd(&cnt[head[e]], 1.0f);
}

// dis = softmax(disen_att, axis=-1) @ weight   (4x16 @ 16x64 -> 4x64), one block of 64
__global__ void dis_kernel(const float* __restrict__ att, const float* __restrict__ w,
                           float* __restrict__ dis) {
    __shared__ float sm[4][16];
    int t = threadIdx.x;
    if (t < 4) {
        float mx = -1e30f;
        for (int r = 0; r < 16; ++r) mx = fmaxf(mx, att[t * 16 + r]);
        float e[16]; float s = 0.0f;
        for (int r = 0; r < 16; ++r) { e[r] = expf(att[t * 16 + r] - mx); s += e[r]; }
        for (int r = 0; r < 16; ++r) sm[t][r] = e[r] / s;
    }
    __syncthreads();
    for (int f = 0; f < 4; ++f) {
        float acc = 0.0f;
        for (int r = 0; r < 16; ++r) acc += sm[f][r] * w[r * 64 + t];
        dis[f * 64 + t] = acc;
    }
}

// zero-padded f16 copy of latent: rows 0..3 = latent, rows 4..15 = 0 (16x64)
__global__ void pack_latent_kernel(const float* __restrict__ latent, _Float16* __restrict__ latf16) {
    int i = blockIdx.x * blockDim.x + threadIdx.x;
    if (i < 16 * 64) latf16[i] = (i < 4 * 64) ? (_Float16)latent[i] : (_Float16)0.0f;
}

// cor: sum of pairwise distance correlations of rows of disen_att (4x16)
__global__ void cor_kernel(const float* __restrict__ att, float* __restrict__ out) {
    if (threadIdx.x != 0 || blockIdx.x != 0) return;
    const int n = 16;
    float total = 0.0f;
    for (int i = 0; i < 4; ++i) {
        for (int j = i + 1; j < 4; ++j) {
            const float* t1 = att + i * 16;
            const float* t2 = att + j * 16;
            float dx[16][16], dy[16][16];
            for (int a = 0; a < n; ++a)
                for (int b = 0; b < n; ++b) {
                    float q1 = t1[a] * t1[a] - 2.0f * t1[a] * t1[b] + t1[b] * t1[b];
                    float q2 = t2[a] * t2[a] - 2.0f * t2[a] * t2[b] + t2[b] * t2[b];
                    dx[a][b] = sqrtf(fmaxf(q1, 0.0f) + 1e-8f);
                    dy[a][b] = sqrtf(fmaxf(q2, 0.0f) + 1e-8f);
                }
            float rmx[16], cmx[16], rmy[16], cmy[16], amx = 0.0f, amy = 0.0f;
            for (int a = 0; a < n; ++a) { rmx[a] = cmx[a] = rmy[a] = cmy[a] = 0.0f; }
            for (int a = 0; a < n; ++a)
                for (int b = 0; b < n; ++b) {
                    rmx[a] += dx[a][b]; cmx[b] += dx[a][b]; amx += dx[a][b];
                    rmy[a] += dy[a][b]; cmy[b] += dy[a][b]; amy += dy[a][b];
                }
            for (int a = 0; a < n; ++a) { rmx[a] /= n; cmx[a] /= n; rmy[a] /= n; cmy[a] /= n; }
            amx /= (float)(n * n); amy /= (float)(n * n);
            float sAB = 0.0f, sAA = 0.0f, sBB = 0.0f;
            for (int a = 0; a < n; ++a)
                for (int b = 0; b < n; ++b) {
                    float A = dx[a][b] - cmx[b] - rmx[a] + amx;
                    float B = dy[a][b] - cmy[b] - rmy[a] + amy;
                    sAB += A * B; sAA += A * A; sBB += B * B;
                }
            float dcAB = sqrtf(fmaxf(sAB / (float)(n * n), 0.0f) + 1e-8f);
            float dcAA = sqrtf(fmaxf(sAA / (float)(n * n), 0.0f) + 1e-8f);
            float dcBB = sqrtf(fmaxf(sBB / (float)(n * n), 0.0f) + 1e-8f);
            total += dcAB / sqrtf(dcAA * dcBB + 1e-8f);
        }
    }
    *out = total;
}

// ---------------- scatter kernels (memory/atomic bound) ----------------

// entity_agg[head] += entity_emb[tail] * weight[type-1]; one thread per (edge, dim)
__global__ void edge_scatter_kernel(const int* __restrict__ head, const int* __restrict__ tail,
                                    const int* __restrict__ etype,
                                    const float* __restrict__ ent, const float* __restrict__ w,
                                    float* __restrict__ agg, int n_edges) {
    int idx = blockIdx.x * blockDim.x + threadIdx.x;
    int e = idx >> 6;
    if (e >= n_edges) return;
    int d = idx & 63;
    int h = head[e], t = tail[e], r = etype[e] - 1;
    float m = ent[t * 64 + d] * w[r * 64 + d];
    atomicAdd(&agg[h * 64 + d], m);
}

// user_agg[row] += val * entity_emb[col]
__global__ void user_scatter_kernel(const int* __restrict__ irow, const int* __restrict__ icol,
                                    const float* __restrict__ ival,
                                    const float* __restrict__ ent,
                                    float* __restrict__ agg, int nnz) {
    int idx = blockIdx.x * blockDim.x + threadIdx.x;
    int e = idx >> 6;
    if (e >= nnz) return;
    int d = idx & 63;
    int r = irow[e], c = icol[e];
    atomicAdd(&agg[r * 64 + d], ival[e] * ent[c * 64 + d]);
}

// ---------------- score GEMM via WMMA ----------------
// logits[u][f] = dot(user_emb[u], latent[f]); 16 users per wave, K=64 via two
// v_wmma_f32_16x16x32_f16 steps; N padded 4->16 via zero-padded f16 latent.
__global__ __launch_bounds__(32) void score_wmma_kernel(const float* __restrict__ user_emb,
                                                        const _Float16* __restrict__ latf16,
                                                        float* __restrict__ score, int n_users) {
    int lane = threadIdx.x;        // 0..31, one full wave
    int u0 = blockIdx.x * 16;
    int m = lane & 15;
    int hi = lane >> 4;            // 0 or 1
    int urow = u0 + m;
    if (urow > n_users - 1) urow = n_users - 1;   // branch-free clamp (stores are guarded)
    const float* rowp = user_emb + urow * 64;
    int n = lane & 15;             // B column

    v8f c = {};
    #pragma unroll
    for (int k0 = 0; k0 < 64; k0 += 32) {
        // A 16x32 f16 layout: lane half selects K-base (+0/+8); halves 0..7 -> K=base+h,
        // halves 8..15 -> K=16+base+h. Vectorized b128 loads, then cvt to f16.
        int abase = k0 + hi * 8;
        float4 x0 = *(const float4*)(rowp + abase);
        float4 x1 = *(const float4*)(rowp + abase + 4);
        float4 y0 = *(const float4*)(rowp + abase + 16);
        float4 y1 = *(const float4*)(rowp + abase + 20);
        v16h a;
        a[0]  = (_Float16)x0.x; a[1]  = (_Float16)x0.y; a[2]  = (_Float16)x0.z; a[3]  = (_Float16)x0.w;
        a[4]  = (_Float16)x1.x; a[5]  = (_Float16)x1.y; a[6]  = (_Float16)x1.z; a[7]  = (_Float16)x1.w;
        a[8]  = (_Float16)y0.x; a[9]  = (_Float16)y0.y; a[10] = (_Float16)y0.z; a[11] = (_Float16)y0.w;
        a[12] = (_Float16)y1.x; a[13] = (_Float16)y1.y; a[14] = (_Float16)y1.z; a[15] = (_Float16)y1.w;

        // B 32x16 f16 layout (matches A): lane half selects K-base (+0/+8); halves 0..7 -> K=base+h,
        // halves 8..15 -> K=16+base+h, col = lane%16. Two b128 loads from the zero-padded latent, no predication.
        int bbase = k0 + hi * 8;
        h8 b0 = *(const h8*)(latf16 + n * 64 + bbase);
        h8 b1 = *(const h8*)(latf16 + n * 64 + bbase + 16);
        v16h b;
        #pragma unroll
        for (int h = 0; h < 8; ++h) { b[h] = b0[h]; b[8 + h] = b1[h]; }

        c = __builtin_amdgcn_wmma_f32_16x16x32_f16(false, a, false, b, (short)0, c, false, false);
    }
    // D layout: VGPR v -> M = v + 8*hi, N = lane%16
    #pragma unroll
    for (int v = 0; v < 8; ++v) {
        int mm = v + hi * 8;
        int nn = lane & 15;
        int u = u0 + mm;
        if (nn < 4 && u < n_users) score[u * 4 + nn] = c[v];
    }
}

__global__ void softmax4_kernel(float* __restrict__ score, int n_users) {
    int u = blockIdx.x * blockDim.x + threadIdx.x;
    if (u >= n_users) return;
    float s0 = score[u * 4 + 0], s1 = score[u * 4 + 1];
    float s2 = score[u * 4 + 2], s3 = score[u * 4 + 3];
    float mx = fmaxf(fmaxf(s0, s1), fmaxf(s2, s3));
    float e0 = expf(s0 - mx), e1 = expf(s1 - mx), e2 = expf(s2 - mx), e3 = expf(s3 - mx);
    float inv = 1.0f / (e0 + e1 + e2 + e3);
    score[u * 4 + 0] = e0 * inv; score[u * 4 + 1] = e1 * inv;
    score[u * 4 + 2] = e2 * inv; score[u * 4 + 3] = e3 * inv;
}

// ---------------- finalize (l2-normalize + residual) ----------------

__global__ __launch_bounds__(64) void entity_final_kernel(const float* __restrict__ agg,
                                                          const float* __restrict__ cnt,
                                                          float* __restrict__ res,
                                                          float* __restrict__ buf) {
    int u = blockIdx.x, d = threadIdx.x;
    float denom = fmaxf(cnt[u], 1.0f);
    float val = agg[u * 64 + d] / denom;
    __shared__ float red[64];
    red[d] = val * val;
    __syncthreads();
    for (int s = 32; s > 0; s >>= 1) { if (d < s) red[d] += red[d + s]; __syncthreads(); }
    float o = val / fmaxf(sqrtf(red[0]), 1e-12f);
    res[u * 64 + d] += o;
    buf[u * 64 + d] = o;
}

__global__ __launch_bounds__(64) void user_final_kernel(const float* __restrict__ agg,
                                                        const float* __restrict__ score,
                                                        const float* __restrict__ dis,
                                                        float* __restrict__ res,
                                                        float* __restrict__ buf) {
    int u = blockIdx.x, d = threadIdx.x;
    float s0 = score[u * 4 + 0], s1 = score[u * 4 + 1];
    float s2 = score[u * 4 + 2], s3 = score[u * 4 + 3];
    float mix = s0 * dis[d] + s1 * dis[64 + d] + s2 * dis[128 + d] + s3 * dis[192 + d];
    float val = agg[u * 64 + d] * (1.0f + mix);
    __shared__ float red[64];
    red[d] = val * val;
    __syncthreads();
    for (int s = 32; s > 0; s >>= 1) { if (d < s) red[d] += red[d + s]; __syncthreads(); }
    float o = val / fmaxf(sqrtf(red[0]), 1e-12f);
    res[u * 64 + d] += o;
    buf[u * 64 + d] = o;
}

// ---------------- launch ----------------

extern "C" void kernel_launch(void* const* d_in, const int* in_sizes, int n_in,
                              void* d_out, int out_size, void* d_ws, size_t ws_size,
                              hipStream_t stream) {
    const float* user_emb   = (const float*)d_in[0];
    const float* entity_emb = (const float*)d_in[1];
    const float* latent     = (const float*)d_in[2];
    const float* weight     = (const float*)d_in[3];
    const float* disen      = (const float*)d_in[4];
    const float* ival       = (const float*)d_in[5];
    const int*   edge_index = (const int*)d_in[6];
    const int*   etype      = (const int*)d_in[7];
    const int*   irow       = (const int*)d_in[8];
    const int*   icol       = (const int*)d_in[9];

    const int* head = edge_index;
    const int* tail = edge_index + N_EDGES;

    float* out        = (float*)d_out;
    float* entity_res = out;
    float* user_res   = out + (size_t)N_ITEMS * DIM;
    float* cor_out    = out + (size_t)out_size - 1;

    float* ws         = (float*)d_ws;
    float* entity_agg = ws;
    float* user_agg   = entity_agg + (size_t)N_ITEMS * DIM;
    float* entity_buf = user_agg   + (size_t)N_USERS * DIM;
    float* user_buf   = entity_buf + (size_t)N_ITEMS * DIM;
    float* cnt        = user_buf   + (size_t)N_USERS * DIM;
    float* score      = cnt + N_ITEMS;
    float* dis        = score + (size_t)N_USERS * 4;
    _Float16* latf16  = (_Float16*)(dis + 4 * DIM);   // 16x64 f16, zero padded

    const int NID = N_ITEMS * DIM;   // 9.6M
    const int NUD = N_USERS * DIM;   // 6.4M

    // constants / init
    zero_f32<<<(N_ITEMS + 255) / 256, 256, 0, stream>>>(cnt, N_ITEMS);
    degree_kernel<<<(N_EDGES + 255) / 256, 256, 0, stream>>>(head, cnt, N_EDGES);
    dis_kernel<<<1, 64, 0, stream>>>(disen, weight, dis);
    pack_latent_kernel<<<4, 256, 0, stream>>>(latent, latf16);
    cor_kernel<<<1, 32, 0, stream>>>(disen, cor_out);
    copy_f32<<<(NID + 255) / 256, 256, 0, stream>>>(entity_emb, entity_res, NID);
    copy_f32<<<(NUD + 255) / 256, 256, 0, stream>>>(user_emb, user_res, NUD);

    const float* ecur = entity_emb;
    const float* ucur = user_emb;
    for (int layer = 0; layer < 2; ++layer) {
        zero_f32<<<(NID + 255) / 256, 256, 0, stream>>>(entity_agg, NID);
        zero_f32<<<(NUD + 255) / 256, 256, 0, stream>>>(user_agg, NUD);

        edge_scatter_kernel<<<(unsigned)(((long long)N_EDGES * 64 + 255) / 256), 256, 0, stream>>>(
            head, tail, etype, ecur, weight, entity_agg, N_EDGES);
        user_scatter_kernel<<<(unsigned)(((long long)NNZ * 64 + 255) / 256), 256, 0, stream>>>(
            irow, icol, ival, ecur, user_agg, NNZ);

        score_wmma_kernel<<<(N_USERS + 15) / 16, 32, 0, stream>>>(ucur, latf16, score, N_USERS);
        softmax4_kernel<<<(N_USERS + 255) / 256, 256, 0, stream>>>(score, N_USERS);

        entity_final_kernel<<<N_ITEMS, 64, 0, stream>>>(entity_agg, cnt, entity_res, entity_buf);
        user_final_kernel<<<N_USERS, 64, 0, stream>>>(user_agg, score, dis, user_res, user_buf);

        ecur = entity_buf;
        ucur = user_buf;
    }
}
